// FNNConv_18236431138965
// MI455X (gfx1250) — matrix operand order repaired and checked by reading, not compiled, not run on hardware
//
#include <hip/hip_runtime.h>
#include <hip/hip_bf16.h>

typedef __attribute__((ext_vector_type(16))) _Float16 v16h;
typedef __attribute__((ext_vector_type(8)))  _Float16 v8h;
typedef __attribute__((ext_vector_type(8)))  float    v8f;

#define BATCH   64
#define SLICES  16
#define NH      512
#define RULES   5184      // 128 blocks * 81 / 2
#define KSPLIT  4
#define KSTEPS  162       // 5184 / 32
#define YSLICE  (BATCH*SLICES*NH)   // 524288

// ---------------------------------------------------------------------------
// Kernel 1: per-(s,h) min/max over batch axis.  8192 threads.
// ---------------------------------------------------------------------------
__global__ __launch_bounds__(256) void minmax_kernel(
    const float* __restrict__ xf, float* __restrict__ mn, float* __restrict__ mx)
{
    const int idx = blockIdx.x * 256 + threadIdx.x;      // 0..8191 = s*512+h
    float lo = xf[idx];
    float hi = lo;
    #pragma unroll 4
    for (int n = 1; n < BATCH; ++n) {
        const float v = xf[n * 8192 + idx];
        lo = fminf(lo, v);
        hi = fmaxf(hi, v);
    }
    mn[idx] = lo;
    mx[idx] = hi;
}

// ---------------------------------------------------------------------------
// Kernel 2: fuzzify + windowed 81-product + normalize + pairwise max -> f16 A.
// One thread per (n, s, block-pair c).  c covers blocks 2c, 2c+1 so the
// pairwise max over the odd-sized (81) rule groups never crosses a thread.
// Normalization uses  sum over 81 products == product of per-window sums.
// ---------------------------------------------------------------------------
__global__ __launch_bounds__(256) void fuzz_kernel(
    const float* __restrict__ xf, const float* __restrict__ C,
    const float* __restrict__ Sigma, const float* __restrict__ mn,
    const float* __restrict__ mx, _Float16* __restrict__ A)
{
    const int idx = blockIdx.x * 256 + threadIdx.x;      // 65536 threads
    const int c  = idx & 63;
    const int s  = (idx >> 6) & 15;
    const int n  = idx >> 10;

    float p32[2][9];   // g3[m3]*g2[m2]  (normalized)
    float p10[2][9];   // g1[m1]*g0[m0]  (normalized)

    #pragma unroll
    for (int sub = 0; sub < 2; ++sub) {
        const int h0 = 8 * c + 4 * sub;                  // block b = 2c+sub, h = 4b..4b+3
        float gn[4][3];
        #pragma unroll
        for (int w = 0; w < 4; ++w) {
            const int h   = h0 + w;
            const int sh  = s * NH + h;
            const float x   = xf[n * 8192 + sh];
            const float mnv = mn[sh];
            const float mxv = mx[sh];
            float z = (x - mnv) / (mxv - mnv);
            if (__builtin_isinf(z)) z = 1.0f;
            float g[3], gs = 0.0f;
            #pragma unroll
            for (int m = 0; m < 3; ++m) {
                const float cm = C[sh * 3 + m];
                const float sg = Sigma[sh * 3 + m];
                const float d  = z - cm;
                g[m] = __expf(-(d * d) / (2.0f * sg));
                gs  += g[m];
            }
            const float inv = 1.0f / gs;
            #pragma unroll
            for (int m = 0; m < 3; ++m) gn[w][m] = g[m] * inv;
        }
        #pragma unroll
        for (int m3 = 0; m3 < 3; ++m3)
            #pragma unroll
            for (int m2 = 0; m2 < 3; ++m2)
                p32[sub][m3 * 3 + m2] = gn[3][m3] * gn[2][m2];
        #pragma unroll
        for (int m1 = 0; m1 < 3; ++m1)
            #pragma unroll
            for (int m0 = 0; m0 < 3; ++m0)
                p10[sub][m1 * 3 + m0] = gn[1][m1] * gn[0][m0];
    }

    // emit 162 values in flat rule order, folding consecutive pairs with max
    _Float16* outp = A + ((n * SLICES + s) * RULES) + c * 81;
    float prev = 0.0f;
    #pragma unroll
    for (int sub = 0; sub < 2; ++sub)
        #pragma unroll
        for (int i = 0; i < 9; ++i)
            #pragma unroll
            for (int j = 0; j < 9; ++j) {
                const int f = sub * 81 + i * 9 + j;      // flat index 0..161
                const float v = p32[sub][i] * p10[sub][j];
                if (f & 1) outp[f >> 1] = (_Float16)fmaxf(prev, v);
                else       prev = v;
            }
}

// ---------------------------------------------------------------------------
// Kernel 3: 16 batched GEMMs  Y_s(64x512) = X_s(64x5184,f16) * W_s(5184x512,f32)
// WMMA f32_16x16x32_f16.  Block = 8 waves -> 64x64 output tile; wave (4x2)
// computes a 16x32 strip (one A frag, two B frags).  K split 4-way over
// blockIdx.z into partial buffers (no atomics -> deterministic).
// coe is converted f32->f16 in-register (single pass over its 170 MB).
// ---------------------------------------------------------------------------
__global__ __launch_bounds__(256) void gemm_kernel(
    const _Float16* __restrict__ A, const float* __restrict__ coe,
    float* __restrict__ ypart)
{
    const int ntile = blockIdx.x;            // 0..7  -> 64 cols of h
    const int s     = blockIdx.y;            // 0..15
    const int kc    = blockIdx.z;            // 0..3
    const int tid   = threadIdx.x;
    const int wave  = tid >> 5;
    const int lane  = tid & 31;
    const int half  = lane >> 4;
    const int l15   = lane & 15;

    const int m0 = (wave & 3) * 16;
    const int n0 = ntile * 64 + (wave >> 2) * 32;

    // A fragment base: row = m0+l15 of slice s; contiguous K order used here (same as B):
    // lanes 0-15 hold K {k0..k0+15}, lanes 16-31 K {k0+16..k0+31}.
    const _Float16* abase = A + ((m0 + l15) * SLICES + s) * RULES + half * 16;
    // B fragment base: B[k][n] = coe[s][k][n]; VGPR i packs K={2i,2i+1}
    // (lanes 0-15) / K={16+2i,17+2i} (lanes 16-31), N striped on l15.
    const float* bbase = coe + (size_t)s * RULES * NH + (size_t)(half * 16) * NH + n0 + l15;

    v8f acc0 = {};
    v8f acc1 = {};

    for (int kk = kc; kk < KSTEPS; kk += KSPLIT) {
        const int k0 = kk * 32;

        union { v16h v; v8h h[2]; } a;
        a.h[0] = *(const v8h*)(abase + k0);        // global_load_b128
        a.h[1] = *(const v8h*)(abase + k0 + 8);    // global_load_b128

        const float* bp = bbase + (size_t)k0 * NH;
        v16h b0 = {}, b1 = {};
        #pragma unroll
        for (int i = 0; i < 8; ++i) {
            const float f00 = bp[(2 * i) * NH];
            const float f01 = bp[(2 * i + 1) * NH];
            const float f10 = bp[(2 * i) * NH + 16];
            const float f11 = bp[(2 * i + 1) * NH + 16];
            b0[2 * i]     = (_Float16)f00;
            b0[2 * i + 1] = (_Float16)f01;
            b1[2 * i]     = (_Float16)f10;
            b1[2 * i + 1] = (_Float16)f11;
        }

        acc0 = __builtin_amdgcn_wmma_f32_16x16x32_f16(
                   false, a.v, false, b0, (short)0, acc0, false, false);
        acc1 = __builtin_amdgcn_wmma_f32_16x16x32_f16(
                   false, a.v, false, b1, (short)0, acc1, false, false);
    }

    // C/D layout: VGPR r -> M = m0 + half*8 + r, N = n0 + l15
    float* yp = ypart + (size_t)kc * YSLICE;
    #pragma unroll
    for (int r = 0; r < 8; ++r) {
        const int m    = m0 + half * 8 + r;
        const int hcol = n0 + l15;
        yp[((m * SLICES + s) << 9) + hcol]      = acc0[r];
        yp[((m * SLICES + s) << 9) + hcol + 16] = acc1[r];
    }
}

// ---------------------------------------------------------------------------
// Kernel 4: reduce 4 K-partials + LayerNorm over (16,512) per sample + PReLU.
// One block per batch sample; y tile cached in 32 KB LDS.
// ---------------------------------------------------------------------------
__global__ __launch_bounds__(256) void lnprelu_kernel(
    const float* __restrict__ ypart, const float* __restrict__ lnw,
    const float* __restrict__ lnb, const float* __restrict__ pa,
    float* __restrict__ out)
{
    __shared__ float ybuf[8192];
    __shared__ float rsum[256];
    __shared__ float rsq[256];

    const int n   = blockIdx.x;
    const int tid = threadIdx.x;
    const float* y0 = ypart + (size_t)n * 8192;

    float s = 0.0f, q = 0.0f;
    for (int i = tid; i < 8192; i += 256) {
        const float v = y0[i] + y0[i + YSLICE] + y0[i + 2 * YSLICE] + y0[i + 3 * YSLICE];
        ybuf[i] = v;
        s += v;
        q += v * v;
    }
    rsum[tid] = s;
    rsq[tid]  = q;
    __syncthreads();
    for (int off = 128; off > 0; off >>= 1) {
        if (tid < off) {
            rsum[tid] += rsum[tid + off];
            rsq[tid]  += rsq[tid + off];
        }
        __syncthreads();
    }
    const float mu  = rsum[0] * (1.0f / 8192.0f);
    const float var = rsq[0] * (1.0f / 8192.0f) - mu * mu;
    const float inv = rsqrtf(var + 1e-5f);
    const float a   = pa[0];

    for (int i = tid; i < 8192; i += 256) {
        float v = (ybuf[i] - mu) * inv * lnw[i] + lnb[i];
        out[(size_t)n * 8192 + i] = (v > 0.0f) ? v : a * v;
    }
}

// ---------------------------------------------------------------------------
extern "C" void kernel_launch(void* const* d_in, const int* in_sizes, int n_in,
                              void* d_out, int out_size, void* d_ws, size_t ws_size,
                              hipStream_t stream)
{
    const float* xf    = (const float*)d_in[0];
    const float* C     = (const float*)d_in[1];
    const float* Sigma = (const float*)d_in[2];
    const float* coe   = (const float*)d_in[3];
    const float* lnw   = (const float*)d_in[4];
    const float* lnb   = (const float*)d_in[5];
    const float* pa    = (const float*)d_in[6];
    float* out = (float*)d_out;

    char* ws = (char*)d_ws;
    float*     mn    = (float*)ws;                         //  32 KB
    float*     mx    = (float*)(ws + 32768);               //  32 KB
    _Float16*  A     = (_Float16*)(ws + 65536);            // 10.6 MB
    float*     ypart = (float*)(ws + 10682368);            //  8.4 MB (4 partials)

    minmax_kernel<<<32, 256, 0, stream>>>(xf, mn, mx);
    fuzz_kernel<<<256, 256, 0, stream>>>(xf, C, Sigma, mn, mx, A);
    dim3 ggrid(8, 16, 4);
    gemm_kernel<<<ggrid, 256, 0, stream>>>(A, coe, ypart);
    lnprelu_kernel<<<64, 256, 0, stream>>>(ypart, lnw, lnb, pa, out);
}
